// GCN_AVG_KHop_Anchored_29643864277065
// MI455X (gfx1250) — matrix operand order repaired and checked by reading, not compiled, not run on hardware
//
#include <hip/hip_runtime.h>
#include <hip/hip_bf16.h>

typedef __bf16 bf16;
typedef __attribute__((ext_vector_type(16))) __bf16 v16bf;
typedef __attribute__((ext_vector_type(8)))  float  v8f;

#define N_IN   256
#define N_HID  128
#define N_OUT  64

// ------------------------------------------------------------------
// elementwise / setup kernels
// ------------------------------------------------------------------
__global__ void k_zero(float* __restrict__ p, long n) {
  long i = (long)blockIdx.x * blockDim.x + threadIdx.x;
  long stride = (long)gridDim.x * blockDim.x;
  for (; i < n; i += stride) p[i] = 0.0f;
}

__global__ void k_count(const int* __restrict__ src, const int* __restrict__ dst,
                        float* __restrict__ deg, float* __restrict__ rowcnt, int E) {
  int e = blockIdx.x * blockDim.x + threadIdx.x;
  if (e >= E) return;
  atomicAdd(&deg[dst[e]], 1.0f);
  atomicAdd(&rowcnt[src[e]], 1.0f);
}

__global__ void k_finalize(const float* __restrict__ deg, const float* __restrict__ rowcnt,
                           float* __restrict__ dinv, float* __restrict__ rowinv, int N) {
  int i = blockIdx.x * blockDim.x + threadIdx.x;
  if (i >= N) return;
  dinv[i] = rsqrtf(deg[i] + 1.0f);           // self-loop => deg >= 1
  float r = rowcnt[i];
  rowinv[i] = (r > 0.0f) ? (1.0f / r) : 1.0f;
}

// Pre-swizzle weights [K x NC] (row-major, K-major) into WMMA B-fragment order:
// Wp[(kt*NT + nt)*512 + lane*16 + e] = W[(kt*32 + (lane>=16?8:0) + (e&7) + (e>=8?16:0))*NC + nt*16 + lane%16]
__global__ void k_pack_w(const float* __restrict__ W, bf16* __restrict__ Wp, int K, int NC) {
  int idx = blockIdx.x * blockDim.x + threadIdx.x;
  int total = (K / 32) * (NC / 16) * 512;
  if (idx >= total) return;
  int e    = idx & 15;
  int lane = (idx >> 4) & 31;
  int t    = idx >> 9;
  int nt   = t % (NC / 16);
  int kt   = t / (NC / 16);
  int k = kt * 32 + ((lane >> 4) ? 8 : 0) + (e & 7) + ((e >> 3) ? 16 : 0);
  int n = nt * 16 + (lane & 15);
  Wp[idx] = (bf16)W[(size_t)k * NC + n];
}

__global__ void k_selfloop(const float* __restrict__ Hsrc, const float* __restrict__ dinv,
                           float* __restrict__ Acc, long total, int cshift) {
  long idx = (long)blockIdx.x * blockDim.x + threadIdx.x;
  if (idx >= total) return;
  long i = idx >> cshift;
  float di = dinv[i];
  Acc[idx] = Hsrc[idx] * di * di;
}

__global__ void k_bias_relu(float* __restrict__ H, const float* __restrict__ b,
                            long total, int cmask) {
  long idx = (long)blockIdx.x * blockDim.x + threadIdx.x;
  if (idx >= total) return;
  float v = H[idx] + b[idx & cmask];
  H[idx] = v > 0.0f ? v : 0.0f;
}

__global__ void k_scale_rows(float* __restrict__ Kb, const float* __restrict__ rowinv,
                             long total, int cshift) {
  long idx = (long)blockIdx.x * blockDim.x + threadIdx.x;
  if (idx >= total) return;
  Kb[idx] *= rowinv[idx >> cshift];
}

__global__ void k_selfloop_bias(const float* __restrict__ T, const float* __restrict__ dinv,
                                const float* __restrict__ b, float* __restrict__ out,
                                long total, int cshift, int cmask) {
  long idx = (long)blockIdx.x * blockDim.x + threadIdx.x;
  if (idx >= total) return;
  long i = idx >> cshift;
  float di = dinv[i];
  out[idx] = T[idx] * di * di + b[idx & cmask];
}

// Acc[sca[e]] += Hsrc[gat[e]] * dinv[gat]*dinv[sca]   (C channels, 4 per thread)
__global__ void k_scatter_norm(const float* __restrict__ Hsrc, float* __restrict__ Acc,
                               const int* __restrict__ gat, const int* __restrict__ sca,
                               const float* __restrict__ dinv, int E, int cshift) {
  long tid = (long)blockIdx.x * blockDim.x + threadIdx.x;
  int tshift = cshift - 2;                 // threads per edge = C/4
  long e = tid >> tshift;
  if (e >= E) return;
  int c4 = (int)(tid & ((1 << tshift) - 1)) << 2;
  int g = gat[e], s = sca[e];
  float nrm = dinv[g] * dinv[s];
  const float4 v = *(const float4*)(Hsrc + ((size_t)g << cshift) + c4);
  float* o = Acc + ((size_t)s << cshift) + c4;
  atomicAdd(o + 0, v.x * nrm);
  atomicAdd(o + 1, v.y * nrm);
  atomicAdd(o + 2, v.z * nrm);
  atomicAdd(o + 3, v.w * nrm);
}

// Acc[sca[e]] += Hsrc[gat[e]]  (un-normalized; used for k-hop numerator)
__global__ void k_scatter_plain(const float* __restrict__ Hsrc, float* __restrict__ Acc,
                                const int* __restrict__ gat, const int* __restrict__ sca,
                                int E, int cshift) {
  long tid = (long)blockIdx.x * blockDim.x + threadIdx.x;
  int tshift = cshift - 2;
  long e = tid >> tshift;
  if (e >= E) return;
  int c4 = (int)(tid & ((1 << tshift) - 1)) << 2;
  int g = gat[e], s = sca[e];
  const float4 v = *(const float4*)(Hsrc + ((size_t)g << cshift) + c4);
  float* o = Acc + ((size_t)s << cshift) + c4;
  atomicAdd(o + 0, v.x);
  atomicAdd(o + 1, v.y);
  atomicAdd(o + 2, v.z);
  atomicAdd(o + 3, v.w);
}

// ------------------------------------------------------------------
// GEMM1: Y[N x 128] = bf16(X[N x 256]) @ W1p    (WMMA bf16, f32 acc)
// block = 256 thr = 8 waves; wave tile 16x16; block tile 32 rows x 64 cols
// ------------------------------------------------------------------
__global__ __launch_bounds__(256)
void k_gemm1_wmma(const float* __restrict__ X, const bf16* __restrict__ Wp,
                  float* __restrict__ Y, int N) {
  const int lane = threadIdx.x & 31;
  const int wave = threadIdx.x >> 5;
  const int lrow = lane & 15;
  const int hi   = lane >> 4;
  const int wr = wave >> 2, wc = wave & 3;
  const int m0 = (blockIdx.x * 2 + wr) * 16;
  const int n0 = (blockIdx.y * 4 + wc) * 16;
  const int nt = n0 >> 4;                  // column tile index (of 8)

  int arow = m0 + lrow;
  if (arow >= N) arow = N - 1;             // clamp; store is guarded
  const float* __restrict__ xr = X + (size_t)arow * N_IN;
  const int off = hi ? 8 : 0;

  v8f c0 = {}; v8f c1 = {};
#pragma unroll
  for (int kt = 0; kt < 8; ++kt) {
    const int k0 = kt * 32;
    // A fragment: runs [k0+off .. +7] and [k0+16+off .. +7]
    const float4 p0 = *(const float4*)(xr + k0 + off);
    const float4 p1 = *(const float4*)(xr + k0 + off + 4);
    const float4 q0 = *(const float4*)(xr + k0 + 16 + off);
    const float4 q1 = *(const float4*)(xr + k0 + 16 + off + 4);
    v16bf a;
    a[0]=(bf16)p0.x; a[1]=(bf16)p0.y; a[2]=(bf16)p0.z; a[3]=(bf16)p0.w;
    a[4]=(bf16)p1.x; a[5]=(bf16)p1.y; a[6]=(bf16)p1.z; a[7]=(bf16)p1.w;
    a[8]=(bf16)q0.x; a[9]=(bf16)q0.y; a[10]=(bf16)q0.z; a[11]=(bf16)q0.w;
    a[12]=(bf16)q1.x; a[13]=(bf16)q1.y; a[14]=(bf16)q1.z; a[15]=(bf16)q1.w;
    const v16bf b = *(const v16bf*)(Wp + (((size_t)kt * 8 + nt) * 32 + lane) * 16);
    if (kt & 1)
      c1 = __builtin_amdgcn_wmma_f32_16x16x32_bf16(false, a, false, b, (short)0, c1, false, false);
    else
      c0 = __builtin_amdgcn_wmma_f32_16x16x32_bf16(false, a, false, b, (short)0, c0, false, false);
  }
#pragma unroll
  for (int r = 0; r < 8; ++r) {
    int row = m0 + (hi ? 8 : 0) + r;
    if (row < N) Y[(size_t)row * N_HID + n0 + lrow] = c0[r] + c1[r];
  }
}

// ------------------------------------------------------------------
// GEMM2: T[N x 64] = bf16(z[N x 256]) @ W2p, z = [h - khop | khop] built on the fly
// ------------------------------------------------------------------
__global__ __launch_bounds__(256)
void k_gemm2_wmma(const float* __restrict__ H, const float* __restrict__ Kh,
                  const bf16* __restrict__ Wp, float* __restrict__ T, int N) {
  const int lane = threadIdx.x & 31;
  const int wave = threadIdx.x >> 5;
  const int lrow = lane & 15;
  const int hi   = lane >> 4;
  const int wr = wave >> 2, wc = wave & 3;
  const int m0 = (blockIdx.x * 2 + wr) * 16;
  const int n0 = wc * 16;                  // 4 column tiles cover all 64 cols

  int arow = m0 + lrow;
  if (arow >= N) arow = N - 1;
  const float* __restrict__ Hr = H  + (size_t)arow * N_HID;
  const float* __restrict__ Kr = Kh + (size_t)arow * N_HID;
  const int off = hi ? 8 : 0;

  v8f c0 = {}; v8f c1 = {};
#pragma unroll
  for (int kt = 0; kt < 8; ++kt) {
    v16bf a;
#pragma unroll
    for (int half = 0; half < 2; ++half) {
      const int kk   = kt * 32 + half * 16 + off;   // compile-time after unroll
      const int base = half * 8;
      if (kk < 128) {                               // z = h - khop
        const float4 h0 = *(const float4*)(Hr + kk);
        const float4 h1 = *(const float4*)(Hr + kk + 4);
        const float4 q0 = *(const float4*)(Kr + kk);
        const float4 q1 = *(const float4*)(Kr + kk + 4);
        a[base+0]=(bf16)(h0.x-q0.x); a[base+1]=(bf16)(h0.y-q0.y);
        a[base+2]=(bf16)(h0.z-q0.z); a[base+3]=(bf16)(h0.w-q0.w);
        a[base+4]=(bf16)(h1.x-q1.x); a[base+5]=(bf16)(h1.y-q1.y);
        a[base+6]=(bf16)(h1.z-q1.z); a[base+7]=(bf16)(h1.w-q1.w);
      } else {                                      // z = khop
        const int kk2 = kk - 128;
        const float4 q0 = *(const float4*)(Kr + kk2);
        const float4 q1 = *(const float4*)(Kr + kk2 + 4);
        a[base+0]=(bf16)q0.x; a[base+1]=(bf16)q0.y;
        a[base+2]=(bf16)q0.z; a[base+3]=(bf16)q0.w;
        a[base+4]=(bf16)q1.x; a[base+5]=(bf16)q1.y;
        a[base+6]=(bf16)q1.z; a[base+7]=(bf16)q1.w;
      }
    }
    const v16bf b = *(const v16bf*)(Wp + (((size_t)kt * 4 + wc) * 32 + lane) * 16);
    if (kt & 1)
      c1 = __builtin_amdgcn_wmma_f32_16x16x32_bf16(false, a, false, b, (short)0, c1, false, false);
    else
      c0 = __builtin_amdgcn_wmma_f32_16x16x32_bf16(false, a, false, b, (short)0, c0, false, false);
  }
#pragma unroll
  for (int r = 0; r < 8; ++r) {
    int row = m0 + (hi ? 8 : 0) + r;
    if (row < N) T[(size_t)row * N_OUT + n0 + lrow] = c0[r] + c1[r];
  }
}

// ------------------------------------------------------------------
// launch
// ------------------------------------------------------------------
extern "C" void kernel_launch(void* const* d_in, const int* in_sizes, int n_in,
                              void* d_out, int out_size, void* d_ws, size_t ws_size,
                              hipStream_t stream) {
  const float* x  = (const float*)d_in[0];
  const float* W1 = (const float*)d_in[1];
  const float* b1 = (const float*)d_in[2];
  const float* W2 = (const float*)d_in[3];
  const float* b2 = (const float*)d_in[4];
  const int*   ei = (const int*)d_in[5];

  const int N = in_sizes[0] / N_IN;
  const int E = in_sizes[5] / 2;
  const int* src = ei;         // edge_index[0]
  const int* dst = ei + E;     // edge_index[1]

  size_t off = 0;
  auto balloc = [&](size_t bytes) -> void* {
    void* p = (void*)((char*)d_ws + off);
    off = (off + bytes + 255) & ~(size_t)255;
    return p;
  };
  float* deg    = (float*)balloc((size_t)N * 4);
  float* rowcnt = (float*)balloc((size_t)N * 4);
  float* dinv   = (float*)balloc((size_t)N * 4);
  float* rowinv = (float*)balloc((size_t)N * 4);
  float* H0     = (float*)balloc((size_t)N * N_HID * 4);  // xW1; reused as T [N x 64]
  float* Hb     = (float*)balloc((size_t)N * N_HID * 4);  // agg -> h (relu)
  float* Kb     = (float*)balloc((size_t)N * N_HID * 4);  // khop numerator -> khop
  bf16*  W1p    = (bf16*)balloc((size_t)8 * 8 * 512 * 2);
  bf16*  W2p    = (bf16*)balloc((size_t)8 * 4 * 512 * 2);
  float* Tb     = H0;                                     // reuse (H0 dead by then)

  const long nhid_total = (long)N * N_HID;
  const long nout_total = (long)N * N_OUT;

  // 1) zero accumulators that are scatter-only
  k_zero<<<1024, 256, 0, stream>>>(deg, N);
  k_zero<<<1024, 256, 0, stream>>>(rowcnt, N);
  k_zero<<<4096, 256, 0, stream>>>(Kb, nhid_total);

  // 2) degree / row counts, 3) norms
  k_count<<<(E + 255) / 256, 256, 0, stream>>>(src, dst, deg, rowcnt, E);
  k_finalize<<<(N + 255) / 256, 256, 0, stream>>>(deg, rowcnt, dinv, rowinv, N);

  // 4) weight pre-swizzle to WMMA fragment layout
  k_pack_w<<<(8 * 8 * 512 + 255) / 256, 256, 0, stream>>>(W1, W1p, N_IN, N_HID);
  k_pack_w<<<(8 * 4 * 512 + 255) / 256, 256, 0, stream>>>(W2, W2p, 2 * N_HID, N_OUT);

  // 5) GEMM1: H0 = x @ W1
  {
    dim3 g((N + 31) / 32, 2);
    k_gemm1_wmma<<<g, 256, 0, stream>>>(x, W1p, H0, N);
  }

  // 6) self-loop init, 7) edge scatter (symmetric norm), 8) bias + relu
  k_selfloop<<<(int)((nhid_total + 255) / 256), 256, 0, stream>>>(H0, dinv, Hb, nhid_total, 7);
  k_scatter_norm<<<(int)(((long)E * 32 + 255) / 256), 256, 0, stream>>>(H0, Hb, src, dst, dinv, E, 7);
  k_bias_relu<<<(int)((nhid_total + 255) / 256), 256, 0, stream>>>(Hb, b1, nhid_total, N_HID - 1);

  // 9) khop numerator: Kb[src] += Hb[dst]; 10) divide by out-degree
  k_scatter_plain<<<(int)(((long)E * 32 + 255) / 256), 256, 0, stream>>>(Hb, Kb, dst, src, E, 7);
  k_scale_rows<<<(int)((nhid_total + 255) / 256), 256, 0, stream>>>(Kb, rowinv, nhid_total, 7);

  // 11) GEMM2: Tb = [h-khop | khop] @ W2   (z fused in fragment loader)
  {
    dim3 g((N + 31) / 32, 1);
    k_gemm2_wmma<<<g, 256, 0, stream>>>(Hb, Kb, W2p, Tb, N);
  }

  // 12) out = Tb*dinv^2 + b2 (self loop + bias), 13) edge scatter into out
  float* out = (float*)d_out;
  k_selfloop_bias<<<(int)((nout_total + 255) / 256), 256, 0, stream>>>(Tb, dinv, b2, out, nout_total, 6, N_OUT - 1);
  k_scatter_norm<<<(int)(((long)E * 16 + 255) / 256), 256, 0, stream>>>(Tb, out, src, dst, dinv, E, 6);
}
